// Model_89309549953143
// MI455X (gfx1250) — matrix operand order repaired and checked by reading, not compiled, not run on hardware
//
#include <hip/hip_runtime.h>
#include <hip/hip_bf16.h>

// ---------------------------------------------------------------------------
// PCNN relation-extraction pipeline for MI455X (gfx1250, wave32, WMMA).
// Dominant cost: the (3x60) conv over 8192x120 token embeddings (~80 GFLOP)
// -> mapped onto v_wmma_f32_16x16x32_bf16 as a per-sentence [128x192]x[192x240]
// GEMM with a masked position-max fused on the accumulator fragments.
// All tables/weights are L2-resident (192 MB L2); HBM traffic ~ a few MB.
// ---------------------------------------------------------------------------

typedef __attribute__((ext_vector_type(16))) __bf16 v16bf;
typedef __attribute__((ext_vector_type(8)))  float  v8f;

#define NSENT 8192
#define LSEQ  120
#define WIDTH 60      // 50 word + 2*5 position
#define DF    230     // conv filters
#define NC    53      // classes
#define KPAD  192     // 3 window rows * 64 (width padded 60->64)
#define NPAD  240     // filters padded 230->240 (15 N-tiles)
#define TPOS  118     // valid conv positions (L-2)

// ---------------------------------------------------------------------------
// K0: pad conv weights (230,3,60) fp32 -> (240,192) bf16, filter-major so a
// B fragment (K=32 slice, per-lane 16 contiguous bf16) is one 32B load.
// ---------------------------------------------------------------------------
__global__ void pad_weights_kernel(const float* __restrict__ Wc,
                                   __bf16* __restrict__ Wcp) {
    int idx = blockIdx.x * 256 + threadIdx.x;
    if (idx >= NPAD * KPAD) return;
    int n = idx / KPAD, k = idx % KPAD;
    int r = k >> 6, c = k & 63;
    float v = 0.0f;
    if (n < DF && c < WIDTH) v = Wc[(n * 3 + r) * WIDTH + c];
    Wcp[idx] = (__bf16)v;
}

// ---------------------------------------------------------------------------
// K1: fused embedding-gather + conv GEMM (WMMA bf16) + masked position-max
//     + tanh  -> sent[N][230].
// One 256-thread block (8 waves) per sentence. Wave w owns M-tile w.
// ---------------------------------------------------------------------------
__global__ void __launch_bounds__(256)
conv_sent_kernel(const int* __restrict__ x, const int* __restrict__ ldist,
                 const int* __restrict__ rdist,
                 const float* __restrict__ Wv, const float* __restrict__ pf1,
                 const float* __restrict__ pf2,
                 const __bf16* __restrict__ Wcp, const float* __restrict__ bc,
                 float* __restrict__ sent) {
    __shared__ __bf16 embS[132 * 64];   // 120 real rows + zero pad, width 64
    __shared__ float  maxS[8 * 240];    // per-wave partial max per filter

    const int s   = blockIdx.x;
    const int tid = threadIdx.x;

    // ---- gather embeddings into LDS, convert fp32 -> bf16 (8448/256 = 33/thr)
    for (int idx = tid; idx < 132 * 64; idx += 256) {
        int row = idx >> 6, col = idx & 63;
        float v = 0.0f;
        if (row < LSEQ && col < WIDTH) {
            int p = s * LSEQ + row;
            if (col < 50)       v = Wv[(long)x[p] * 50 + col];
            else if (col < 55)  v = pf1[ldist[p] * 5 + (col - 50)];
            else                v = pf2[rdist[p] * 5 + (col - 55)];
        }
        embS[idx] = (__bf16)v;
    }
    __syncthreads();

    const int wave = tid >> 5;   // == M-tile (t in [wave*16, wave*16+16))
    const int lane = tid & 31;
    const int m    = lane & 15;
    const int hi   = lane >> 4;

    // ---- preload A fragments for all 6 K-steps (contiguous k order, matching B):
    // element e -> klocal = hi*16 + e
    v16bf a[6];
#pragma unroll
    for (int kk = 0; kk < 6; ++kk) {
#pragma unroll
        for (int e = 0; e < 16; ++e) {
            int vv = e >> 1, p = e & 1;
            int klocal = hi * 16 + e; (void)vv; (void)p;
            int kg  = kk * 32 + klocal;        // 0..191
            int r   = kg >> 6, col = kg & 63;  // window row / padded width col
            a[kk][e] = embS[(wave * 16 + m + r) * 64 + col];
        }
    }

    // ---- sweep 15 N-tiles, reusing A; B frag = 16 contiguous bf16 per lane
    for (int nt = 0; nt < 15; ++nt) {
        v8f acc = {0.f, 0.f, 0.f, 0.f, 0.f, 0.f, 0.f, 0.f};
        const __bf16* wp = Wcp + ((size_t)(nt * 16 + m) * KPAD + hi * 16);
#pragma unroll
        for (int kk = 0; kk < 6; ++kk) {
            v16bf b = *(const v16bf*)(wp + kk * 32);   // 32B aligned
            acc = __builtin_amdgcn_wmma_f32_16x16x32_bf16(
                false, a[kk], false, b, (short)0, acc, false, false);
        }
        // masked max over this tile's 16 positions (D layout: M = i + hi*8)
        float lm = -3.0e38f;
#pragma unroll
        for (int i = 0; i < 8; ++i) {
            int t = wave * 16 + i + hi * 8;
            lm = fmaxf(lm, (t < TPOS) ? acc[i] : -3.0e38f);
        }
        lm = fmaxf(lm, __shfl_xor(lm, 16, 32));        // merge lane halves
        if (lane < 16) maxS[wave * 240 + nt * 16 + lane] = lm;
    }
    __syncthreads();

    // ---- reduce across 8 waves (M-tiles), add bias, tanh, store sentence rep
    for (int d = tid; d < DF; d += 256) {
        float mx = maxS[d];
#pragma unroll
        for (int w = 1; w < 8; ++w) mx = fmaxf(mx, maxS[w * 240 + d]);
        sent[(long)s * DF + d] = tanhf(mx + bc[d]);
    }
}

// ---------------------------------------------------------------------------
// K2: e[N,C] = sent @ Wr^T + br   (0.2 GFLOP, scalar is fine)
// ---------------------------------------------------------------------------
__global__ void logits_kernel(const float* __restrict__ sent,
                              const float* __restrict__ Wr,
                              const float* __restrict__ br,
                              float* __restrict__ eo, int n) {
    int j = blockIdx.x * 256 + threadIdx.x;
    if (j >= n * NC) return;
    int i = j / NC, c = j % NC;
    const float* sp = sent + (long)i * DF;
    const float* wp = Wr + c * DF;
    float acc = br[c];
    for (int d = 0; d < DF; ++d) acc += sp[d] * wp[d];
    eo[j] = acc;
}

// ---------------------------------------------------------------------------
// K3: ragged per-bag softmax over sentences (per class) honoring total_shape.
// One block per bag; thread c handles class c (3 passes over <=32 sentences).
// ---------------------------------------------------------------------------
__global__ void alpha_kernel(const float* __restrict__ eo,
                             const int* __restrict__ ts,
                             float* __restrict__ alpha) {
    int g = blockIdx.x;
    int c = threadIdx.x;
    if (c >= NC) return;
    int s0 = ts[g], s1 = ts[g + 1];
    float mx = -3.0e38f;
    for (int i = s0; i < s1; ++i) mx = fmaxf(mx, eo[i * NC + c]);
    float den = 0.0f;
    for (int i = s0; i < s1; ++i) den += expf(eo[i * NC + c] - mx);
    for (int i = s0; i < s1; ++i)
        alpha[i * NC + c] = expf(eo[i * NC + c] - mx) / den;
}

// ---------------------------------------------------------------------------
// K4: per (bag g, class c): bagvec = sum_i alpha[i,c]*sent[i,:],
//     scores[k] = bagvec . Wr[k] + br[k], softmax over k, emit diagonal.
// ---------------------------------------------------------------------------
__global__ void __launch_bounds__(256)
bag_kernel(const float* __restrict__ sent, const float* __restrict__ alpha,
           const int* __restrict__ ts, const float* __restrict__ Wr,
           const float* __restrict__ br, float* __restrict__ out) {
    __shared__ float bagv[DF];
    __shared__ float sc[NC];
    int g = blockIdx.x / NC, c = blockIdx.x % NC;
    int tid = threadIdx.x;
    int s0 = ts[g], s1 = ts[g + 1];

    for (int d = tid; d < DF; d += 256) {
        float acc = 0.0f;
        for (int i = s0; i < s1; ++i)
            acc += alpha[i * NC + c] * sent[(long)i * DF + d];
        bagv[d] = acc;
    }
    __syncthreads();

    for (int k = tid; k < NC; k += 256) {
        float acc = br[k];
        for (int d = 0; d < DF; ++d) acc += bagv[d] * Wr[k * DF + d];
        sc[k] = acc;
    }
    __syncthreads();

    if (tid == 0) {
        float mx = sc[0];
        for (int k = 1; k < NC; ++k) mx = fmaxf(mx, sc[k]);
        float den = 0.0f;
        for (int k = 0; k < NC; ++k) den += expf(sc[k] - mx);
        out[g * NC + c] = expf(sc[c] - mx) / den;
    }
}

// ---------------------------------------------------------------------------
extern "C" void kernel_launch(void* const* d_in, const int* in_sizes, int n_in,
                              void* d_out, int out_size, void* d_ws, size_t ws_size,
                              hipStream_t stream) {
    const int*   x   = (const int*)d_in[0];
    const int*   ld  = (const int*)d_in[1];
    const int*   rd  = (const int*)d_in[2];
    const int*   ts  = (const int*)d_in[3];
    const float* Wv  = (const float*)d_in[4];
    const float* pf1 = (const float*)d_in[5];
    const float* pf2 = (const float*)d_in[6];
    const float* Wc  = (const float*)d_in[7];
    const float* bc  = (const float*)d_in[8];
    const float* Wr  = (const float*)d_in[9];
    const float* br  = (const float*)d_in[10];
    float* out = (float*)d_out;

    const int N = in_sizes[0] / LSEQ;     // 8192
    const int G = in_sizes[3] - 1;        // 256

    // workspace layout (256B-aligned slices): Wcp | sent | e | alpha  (~11 MB)
    char*  ws  = (char*)d_ws;
    size_t off = 0;
    __bf16* Wcp = (__bf16*)(ws + off);
    off += ((size_t)NPAD * KPAD * sizeof(__bf16) + 255) & ~(size_t)255;
    float* sent = (float*)(ws + off);
    off += ((size_t)N * DF * sizeof(float) + 255) & ~(size_t)255;
    float* eo   = (float*)(ws + off);
    off += ((size_t)N * NC * sizeof(float) + 255) & ~(size_t)255;
    float* alpha = (float*)(ws + off);

    pad_weights_kernel<<<(NPAD * KPAD + 255) / 256, 256, 0, stream>>>(Wc, Wcp);
    conv_sent_kernel<<<N, 256, 0, stream>>>(x, ld, rd, Wv, pf1, pf2, Wcp, bc, sent);
    logits_kernel<<<(N * NC + 255) / 256, 256, 0, stream>>>(sent, Wr, br, eo, N);
    alpha_kernel<<<G, 64, 0, stream>>>(eo, ts, alpha);
    bag_kernel<<<G * NC, 256, 0, stream>>>(sent, alpha, ts, Wr, br, out);
}
